// mLSTMLayer_85126251807345
// MI455X (gfx1250) — matrix operand order repaired and checked by reading, not compiled, not run on hardware
//
#include <hip/hip_runtime.h>
#include <hip/hip_bf16.h>

// Problem constants (match reference)
#define BB    2
#define SSEQ  2048
#define EE    1024
#define INNER 2048
#define NHEAD 4
#define DHEAD 512   // INNER / NHEAD

typedef __attribute__((ext_vector_type(16))) __bf16 bf16x16;
typedef __attribute__((ext_vector_type(4)))  __bf16 bf16x4;
typedef __attribute__((ext_vector_type(8)))  float  f32x8;
typedef __attribute__((ext_vector_type(4)))  float  f32x4;

// ---------------------------------------------------------------------------
// WMMA helper: D = A(16x32 bf16) * B(32x16 bf16) + C(16x16 f32)
// ---------------------------------------------------------------------------
__device__ __forceinline__ f32x8 wmma_bf16(bf16x16 a, bf16x16 b, f32x8 c) {
#if defined(__HIP_DEVICE_COMPILE__)
  return __builtin_amdgcn_wmma_f32_16x16x32_bf16(false, a, false, b,
                                                 (short)0, c, false, false);
#else
  return c;
#endif
}

__device__ __forceinline__ bf16x4 cvt4(f32x4 v) {
  bf16x4 r;
  r[0] = (__bf16)v[0]; r[1] = (__bf16)v[1];
  r[2] = (__bf16)v[2]; r[3] = (__bf16)v[3];
  return r;
}

// 16 contiguous f32 (16B-aligned) -> bf16 fragment (B operand)
__device__ __forceinline__ bf16x16 cvt_frag16(const float* __restrict__ p) {
  const f32x4* q = (const f32x4*)p;
  f32x4 a = q[0], b = q[1], c = q[2], d = q[3];
  bf16x16 r;
#pragma unroll
  for (int i = 0; i < 4; ++i) {
    r[i]      = (__bf16)a[i];
    r[i + 4]  = (__bf16)b[i];
    r[i + 8]  = (__bf16)c[i];
    r[i + 12] = (__bf16)d[i];
  }
  return r;
}
// two contiguous 8-f32 chunks (16B-aligned) -> bf16 fragment (A operand)
__device__ __forceinline__ bf16x16 cvt_frag8x2(const float* __restrict__ p0,
                                               const float* __restrict__ p1) {
  const f32x4* q0 = (const f32x4*)p0;
  const f32x4* q1 = (const f32x4*)p1;
  f32x4 a = q0[0], b = q0[1], c = q1[0], d = q1[1];
  bf16x16 r;
#pragma unroll
  for (int i = 0; i < 4; ++i) {
    r[i]      = (__bf16)a[i];
    r[i + 4]  = (__bf16)b[i];
    r[i + 8]  = (__bf16)c[i];
    r[i + 12] = (__bf16)d[i];
  }
  return r;
}
// two contiguous 8-bf16 chunks from LDS -> fragment
__device__ __forceinline__ bf16x16 lds_frag8x2(const __bf16* p0, const __bf16* p1) {
  bf16x16 r;
#pragma unroll
  for (int i = 0; i < 8; ++i) { r[i] = p0[i]; r[i + 8] = p1[i]; }
  return r;
}

__device__ __forceinline__ f32x8 zero8() {
  f32x8 z = {0.f, 0.f, 0.f, 0.f, 0.f, 0.f, 0.f, 0.f};
  return z;
}
__device__ __forceinline__ void load8(const float* __restrict__ p, float* d) {
  const f32x4* q = (const f32x4*)p;
  f32x4 a = q[0], b = q[1];
#pragma unroll
  for (int i = 0; i < 4; ++i) { d[i] = a[i]; d[i + 4] = b[i]; }
}
__device__ __forceinline__ void store8(float* __restrict__ p, const float* d) {
  f32x4 a, b;
#pragma unroll
  for (int i = 0; i < 4; ++i) { a[i] = d[i]; b[i] = d[i + 4]; }
  f32x4* q = (f32x4*)p;
  q[0] = a; q[1] = b;
}

// ---------------------------------------------------------------------------
// Kernel 1/6: tiled bf16-WMMA GEMM  C = A(f32,MxK) * B(f32,KxN)
// 128x128 tile per 256-thread block, BK=64; wave w owns rows 16w..16w+15.
// B fragments batched into distinct registers so DS loads overlap WMMAs.
// Output columns < splitN go to C0 (row stride splitN), rest to C1.
// ---------------------------------------------------------------------------
__global__ __launch_bounds__(256) void gemm_bf16_kernel(
    const float* __restrict__ A, const float* __restrict__ Bw,
    float* __restrict__ C0, float* __restrict__ C1,
    int M, int K, int N, int splitN) {
  __shared__ alignas(16) __bf16 As[128][72];   // [m][k], BK=64 + pad
  __shared__ alignas(16) __bf16 Bs[128][72];   // [n][k] (transposed stage)

  const int tid  = threadIdx.x;
  const int wave = tid >> 5, lane = tid & 31;
  const int ml   = lane & 15, half = lane >> 4;
  const int m0 = blockIdx.y * 128, n0 = blockIdx.x * 128;

  f32x8 acc[8];
#pragma unroll
  for (int nt = 0; nt < 8; ++nt) acc[nt] = zero8();

  for (int k0 = 0; k0 < K; k0 += 64) {
    __syncthreads();
    // ---- A tile: 128x64, 8 float4 per thread, batched loads ----
    f32x4 av[8];
#pragma unroll
    for (int i = 0; i < 8; ++i) {
      int q = i * 256 + tid;
      int r = q >> 4, kq = q & 15;
      av[i] = *(const f32x4*)&A[(size_t)(m0 + r) * K + (k0 + kq * 4)];
    }
#pragma unroll
    for (int i = 0; i < 8; ++i) {
      int q = i * 256 + tid;
      int r = q >> 4, kq = q & 15;
      *(bf16x4*)&As[r][kq * 4] = cvt4(av[i]);
    }
    // ---- B tile: 64x128 -> Bs[n][k], 8 float4 per thread ----
    f32x4 bv[8];
#pragma unroll
    for (int i = 0; i < 8; ++i) {
      int q = i * 256 + tid;
      int kk = q >> 5, nq = q & 31;
      bv[i] = *(const f32x4*)&Bw[(size_t)(k0 + kk) * N + (n0 + nq * 4)];
    }
#pragma unroll
    for (int i = 0; i < 8; ++i) {
      int q = i * 256 + tid;
      int kk = q >> 5, nq = q & 31;
      bf16x4 b4 = cvt4(bv[i]);
#pragma unroll
      for (int j = 0; j < 4; ++j) Bs[nq * 4 + j][kk] = b4[j];
    }
    if (k0 + 64 < K)                          // CDNA5 global_prefetch_b8 path
      __builtin_prefetch(&A[(size_t)(m0 + (tid >> 1)) * K + k0 + 64 + ((tid & 1) << 5)], 0, 1);
    __syncthreads();

    const __bf16* arow = &As[wave * 16 + ml][0];
    bf16x16 af0 = lds_frag8x2(arow + 8 * half,      arow + 16 + 8 * half);
    bf16x16 af1 = lds_frag8x2(arow + 32 + 8 * half, arow + 48 + 8 * half);

    bf16x16 bfr[8];
#pragma unroll
    for (int nt = 0; nt < 8; ++nt) {          // batch: 8 frag loads in flight
      const __bf16* brow = &Bs[nt * 16 + ml][0];
      bfr[nt] = lds_frag8x2(brow + 8 * half, brow + 16 + 8 * half);
    }
#pragma unroll
    for (int nt = 0; nt < 8; ++nt) acc[nt] = wmma_bf16(af0, bfr[nt], acc[nt]);
#pragma unroll
    for (int nt = 0; nt < 8; ++nt) {
      const __bf16* brow = &Bs[nt * 16 + ml][0];
      bfr[nt] = lds_frag8x2(brow + 32 + 8 * half, brow + 48 + 8 * half);
    }
#pragma unroll
    for (int nt = 0; nt < 8; ++nt) acc[nt] = wmma_bf16(af1, bfr[nt], acc[nt]);
  }

  // store (C/D layout: lane holds col n=ml, rows r + 8*half)
#pragma unroll
  for (int nt = 0; nt < 8; ++nt) {
    int col = n0 + nt * 16 + ml;
    float* dst; int cc;
    if (col < splitN) { dst = C0; cc = col; } else { dst = C1; cc = col - splitN; }
#pragma unroll
    for (int r = 0; r < 8; ++r) {
      int row = m0 + wave * 16 + r + 8 * half;
      dst[(size_t)row * splitN + cc] = acc[nt][r];
    }
  }
}

// ---------------------------------------------------------------------------
// Kernel 2: causal depthwise conv + swish, headwise q/k/v (4x4 blocks),
// gate preactivations ig/fg.  One block per token (b,s), 256 thr x 8 channels.
// ---------------------------------------------------------------------------
__global__ __launch_bounds__(256) void pre_kernel(
    const float* __restrict__ xm, const float* __restrict__ conv_w,
    const float* __restrict__ conv_b,
    const float* __restrict__ Wq, const float* __restrict__ Wk,
    const float* __restrict__ Wv,
    const float* __restrict__ W_ig, const float* __restrict__ b_ig,
    const float* __restrict__ W_fg, const float* __restrict__ b_fg,
    float* __restrict__ xa_out, float* __restrict__ q_out,
    float* __restrict__ k_out, float* __restrict__ vt_out,
    float* __restrict__ ig_out, float* __restrict__ fg_out) {
  const int bs = blockIdx.x;
  const int b = bs / SSEQ, s = bs % SSEQ;
  const int tid = threadIdx.x, c0 = tid * 8;

  float a8[8], xmv[8], xav[8];
  load8(&conv_b[c0], a8);
#pragma unroll
  for (int i = 0; i < 4; ++i) {
    int sp = s - 3 + i;
    if (sp >= 0) {
      float xr[8], wr[8];
      load8(&xm[(size_t)(b * SSEQ + sp) * INNER + c0], xr);
      load8(&conv_w[i * INNER + c0], wr);
#pragma unroll
      for (int e = 0; e < 8; ++e) a8[e] += xr[e] * wr[e];
      if (i == 3)
#pragma unroll
        for (int e = 0; e < 8; ++e) xmv[e] = xr[e];
    }
  }
#pragma unroll
  for (int e = 0; e < 8; ++e) xav[e] = a8[e] / (1.f + __expf(-a8[e]));  // swish
  store8(&xa_out[(size_t)bs * INNER + c0], xav);

  float qv[8], kv[8], vv[8];
#pragma unroll
  for (int j = 0; j < 2; ++j) {               // two 4-wide blocks per thread
    int g = (c0 >> 2) + j;
    float wq[16], wk[16], wv[16];
    load8(Wq + (size_t)g * 16, wq); load8(Wq + (size_t)g * 16 + 8, wq + 8);
    load8(Wk + (size_t)g * 16, wk); load8(Wk + (size_t)g * 16 + 8, wk + 8);
    load8(Wv + (size_t)g * 16, wv); load8(Wv + (size_t)g * 16 + 8, wv + 8);
#pragma unroll
    for (int o = 0; o < 4; ++o) {
      float aq = 0.f, ak = 0.f, av = 0.f;
#pragma unroll
      for (int d = 0; d < 4; ++d) {
        aq += wq[o * 4 + d] * xav[j * 4 + d];
        ak += wk[o * 4 + d] * xav[j * 4 + d];
        av += wv[o * 4 + d] * xmv[j * 4 + d];
      }
      qv[j * 4 + o] = aq; kv[j * 4 + o] = ak; vv[j * 4 + o] = av;
    }
  }
  store8(&q_out[(size_t)bs * INNER + c0], qv);
  store8(&k_out[(size_t)bs * INNER + c0], kv);
#pragma unroll
  for (int e = 0; e < 8; ++e)
    vt_out[((size_t)b * INNER + c0 + e) * SSEQ + s] = vv[e];  // d-major for PV

  // gate preactivations: concat(q,k,v) @ W_{ig,fg}  (f32x4 rows)
  float pg[8] = {0, 0, 0, 0, 0, 0, 0, 0};
#pragma unroll
  for (int e = 0; e < 8; ++e) {
    int c = c0 + e;
    f32x4 wiq = *(const f32x4*)&W_ig[(size_t)c * 4];
    f32x4 wik = *(const f32x4*)&W_ig[(size_t)(INNER + c) * 4];
    f32x4 wiv = *(const f32x4*)&W_ig[(size_t)(2 * INNER + c) * 4];
    f32x4 wfq = *(const f32x4*)&W_fg[(size_t)c * 4];
    f32x4 wfk = *(const f32x4*)&W_fg[(size_t)(INNER + c) * 4];
    f32x4 wfv = *(const f32x4*)&W_fg[(size_t)(2 * INNER + c) * 4];
#pragma unroll
    for (int nh = 0; nh < 4; ++nh) {
      pg[nh]     += qv[e] * wiq[nh] + kv[e] * wik[nh] + vv[e] * wiv[nh];
      pg[4 + nh] += qv[e] * wfq[nh] + kv[e] * wfk[nh] + vv[e] * wfv[nh];
    }
  }
  __shared__ float red[8][256];
#pragma unroll
  for (int g = 0; g < 8; ++g) red[g][tid] = pg[g];
  __syncthreads();
  for (int st = 128; st > 0; st >>= 1) {
    if (tid < st)
#pragma unroll
      for (int g = 0; g < 8; ++g) red[g][tid] += red[g][tid + st];
    __syncthreads();
  }
  if (tid < 4)
    ig_out[((size_t)(b * NHEAD + tid)) * SSEQ + s] = red[tid][0] + b_ig[tid];
  else if (tid < 8)
    fg_out[((size_t)(b * NHEAD + (tid - 4))) * SSEQ + s] = red[tid][0] + b_fg[tid - 4];
}

// ---------------------------------------------------------------------------
// Kernel 3: per-(b,nh) gate scan. lfc cumsum of log_sigmoid(fg), plus
//   gcol[t] = ig[t]-lfc[t+1];  rowm[s] = prefix-max(gcol);  mD[s]=lfc[s+1]+rowm[s]
// ---------------------------------------------------------------------------
__global__ void scan_kernel(const float* __restrict__ ig,
                            const float* __restrict__ fg,
                            float* __restrict__ gcol,
                            float* __restrict__ rowm,
                            float* __restrict__ mD) {
  const size_t base = (size_t)blockIdx.x * SSEQ;
  float lf = 0.f, gm = -1e30f;
  for (int t = 0; t < SSEQ; ++t) {
    float f  = fg[base + t];
    float ls = (f >= 0.f) ? -log1pf(__expf(-f)) : f - log1pf(__expf(f));
    lf += ls;
    float gc = ig[base + t] - lf;
    gcol[base + t] = gc;
    gm = fmaxf(gm, gc);
    rowm[base + t] = gm;
    mD[base + t]   = lf + gm;
  }
}

// ---------------------------------------------------------------------------
// Kernel 4: streaming mLSTM attention with WMMA.
// 8 independent waves: qsub = wave>>2 (two 16-row q tiles),
// dq = wave&3 (four 128-col DH quarters). Q fragments resident in VGPRs.
// K fragments batched x4, V fragments batched x8 for load/WMMA overlap.
// ---------------------------------------------------------------------------
__global__ __launch_bounds__(256) void attn_kernel(
    const float* __restrict__ qb, const float* __restrict__ kb,
    const float* __restrict__ vt,
    const float* __restrict__ gcol, const float* __restrict__ rowm,
    const float* __restrict__ mD, float* __restrict__ h) {
  const float scale = 0.044194173824159216f;  // DH^-0.5
  const int tid = threadIdx.x, wave = tid >> 5, lane = tid & 31;
  const int ml = lane & 15, half = lane >> 4;
  const int qsub = wave >> 2, dq = wave & 3;
  const int b = blockIdx.z, nh = blockIdx.y;
  const int q0 = blockIdx.x * 32 + qsub * 16;
  const size_t gbase = (size_t)(b * NHEAD + nh) * SSEQ;

  // resident Q fragments (16 k-steps of 32 over DH=512)
  const float* qrow = qb + (size_t)(b * SSEQ + q0 + ml) * INNER + nh * DHEAD;
  bf16x16 qf[16];
#pragma unroll
  for (int ks = 0; ks < 16; ++ks) {
    int d0 = ks * 32;
    qf[ks] = cvt_frag8x2(qrow + d0 + 16 * half, qrow + d0 + 16 * half + 8);
  }
  float rm[8], em[8];
#pragma unroll
  for (int r = 0; r < 8; ++r) {
    int row = q0 + r + 8 * half;
    rm[r] = rowm[gbase + row];
    em[r] = __expf(-mD[gbase + row]);
  }

  f32x8 acc[8];
#pragma unroll
  for (int nt = 0; nt < 8; ++nt) acc[nt] = zero8();
  float rsum[8] = {0, 0, 0, 0, 0, 0, 0, 0};

  __shared__ alignas(16) __bf16 Cst[8][16][40];   // per-wave C~ staging

  const int ktn = blockIdx.x + 1;                 // key tiles covering t<=s
  for (int kt = 0; kt < ktn; ++kt) {
    int t0 = kt * 32;
    // ---- QK^T (two 16-key subtiles), decay gating, row sums ----
#pragma unroll
    for (int ksub = 0; ksub < 2; ++ksub) {
      int key = t0 + ksub * 16 + ml;
      const float* krow = kb + (size_t)(b * SSEQ + key) * INNER + nh * DHEAD;
      f32x8 c = zero8();
#pragma unroll
      for (int ks4 = 0; ks4 < 4; ++ks4) {         // 4 K-frags in flight
        bf16x16 kf[4];
#pragma unroll
        for (int j = 0; j < 4; ++j)
          kf[j] = cvt_frag16(krow + (ks4 * 4 + j) * 32 + 16 * half);
#pragma unroll
        for (int j = 0; j < 4; ++j)
          c = wmma_bf16(qf[ks4 * 4 + j], kf[j], c);
      }
      float gc = gcol[gbase + key];
#pragma unroll
      for (int r = 0; r < 8; ++r) {
        int srow = q0 + r + 8 * half;
        float val = (key <= srow) ? c[r] * scale * __expf(gc - rm[r]) : 0.f;
        rsum[r] += val;
        Cst[wave][r + 8 * half][ksub * 16 + ml] = (__bf16)val;
      }
    }
#if defined(__HIP_DEVICE_COMPILE__)
    asm volatile("s_wait_dscnt 0" ::: "memory");  // same-wave LDS RAW fence
#endif
    // ---- PV over this wave's DH quarter, 8 V-frags batched ----
    bf16x16 af = lds_frag8x2(&Cst[wave][ml][16 * half], &Cst[wave][ml][16 * half + 8]);
    bf16x16 vf[8];
#pragma unroll
    for (int nt = 0; nt < 8; ++nt) {
      int dc = nh * DHEAD + dq * 128 + nt * 16 + ml;
      const float* vrow = vt + ((size_t)b * INNER + dc) * SSEQ;
      vf[nt] = cvt_frag16(vrow + t0 + 16 * half);
    }
#pragma unroll
    for (int nt = 0; nt < 8; ++nt) acc[nt] = wmma_bf16(af, vf[nt], acc[nt]);
#if defined(__HIP_DEVICE_COMPILE__)
    asm volatile("s_wait_dscnt 0" ::: "memory");  // WAR before next-tile writes
#endif
  }

  // row-sum reduce across the 16 n-lanes of each half
#pragma unroll
  for (int mask = 1; mask < 16; mask <<= 1)
#pragma unroll
    for (int r = 0; r < 8; ++r) rsum[r] += __shfl_xor(rsum[r], mask, 32);

  float inv[8];
#pragma unroll
  for (int r = 0; r < 8; ++r)
    inv[r] = 1.f / (fmaxf(fabsf(rsum[r]), em[r]) + 1e-6f);

#pragma unroll
  for (int nt = 0; nt < 8; ++nt) {
    int dc = nh * DHEAD + dq * 128 + nt * 16 + ml;
#pragma unroll
    for (int r = 0; r < 8; ++r) {
      int row = q0 + r + 8 * half;
      h[(size_t)(b * SSEQ + row) * INNER + dc] = acc[nt][r] * inv[r];
    }
  }
}

// ---------------------------------------------------------------------------
// Kernel 5: multi-head layernorm over DH + skip + swish(z) gating.
// ---------------------------------------------------------------------------
__global__ __launch_bounds__(256) void post_kernel(
    const float* __restrict__ h, const float* __restrict__ xa,
    const float* __restrict__ z, const float* __restrict__ norm_w,
    const float* __restrict__ skip, float* __restrict__ hs) {
  const int bs = blockIdx.x, tid = threadIdx.x, c0 = tid * 8;
  const size_t base = (size_t)bs * INNER;
  float hv[8], sum = 0.f, sq = 0.f;
  load8(&h[base + c0], hv);
#pragma unroll
  for (int e = 0; e < 8; ++e) { sum += hv[e]; sq += hv[e] * hv[e]; }
  __shared__ float s1[256], s2[256];
  s1[tid] = sum; s2[tid] = sq;
  __syncthreads();
  for (int st = 32; st > 0; st >>= 1) {
    if ((tid & 63) < st) { s1[tid] += s1[tid + st]; s2[tid] += s2[tid + st]; }
    __syncthreads();
  }
  const int hb = tid & ~63;
  float mu  = s1[hb] * (1.f / DHEAD);
  float var = s2[hb] * (1.f / DHEAD) - mu * mu;
  float rin = rsqrtf(var + 1e-6f);
  float zv[8], xv[8], nw[8], sk[8], o8[8];
  load8(&z[base + c0], zv);
  load8(&xa[base + c0], xv);
  load8(&norm_w[c0], nw);
  load8(&skip[c0], sk);
#pragma unroll
  for (int e = 0; e < 8; ++e) {
    float hn = (hv[e] - mu) * rin * nw[e];
    float sw = zv[e] / (1.f + __expf(-zv[e]));
    o8[e] = (hn + sk[e] * xv[e]) * sw;
  }
  store8(&hs[base + c0], o8);
}

// ---------------------------------------------------------------------------
// Host launcher
// ---------------------------------------------------------------------------
extern "C" void kernel_launch(void* const* d_in, const int* in_sizes, int n_in,
                              void* d_out, int out_size, void* d_ws, size_t ws_size,
                              hipStream_t stream) {
  (void)in_sizes; (void)n_in; (void)out_size; (void)ws_size;
  const float* x      = (const float*)d_in[0];
  const float* W_up   = (const float*)d_in[1];
  const float* conv_w = (const float*)d_in[2];
  const float* conv_b = (const float*)d_in[3];
  const float* Wq     = (const float*)d_in[4];
  const float* Wk     = (const float*)d_in[5];
  const float* Wv     = (const float*)d_in[6];
  const float* W_ig   = (const float*)d_in[7];
  const float* b_ig   = (const float*)d_in[8];
  const float* W_fg   = (const float*)d_in[9];
  const float* b_fg   = (const float*)d_in[10];
  const float* norm_w = (const float*)d_in[11];
  const float* skip   = (const float*)d_in[12];
  const float* W_down = (const float*)d_in[13];
  float* out = (float*)d_out;

  const size_t BIG = (size_t)BB * SSEQ * INNER;   // 8M floats per buffer
  float* ws = (float*)d_ws;
  float* xm = ws;               // x_mlstm (reused for h_state)
  float* zb = ws + 1 * BIG;
  float* xa = ws + 2 * BIG;
  float* qb = ws + 3 * BIG;
  float* kb = ws + 4 * BIG;
  float* vt = ws + 5 * BIG;     // v transposed, (b, channel, s)
  float* hb = ws + 6 * BIG;
  float* sm = ws + 7 * BIG;     // small gate arrays
  const size_t GS = (size_t)BB * NHEAD * SSEQ;    // 16384
  float* igb  = sm;
  float* fgb  = sm + 1 * GS;
  float* gcol = sm + 2 * GS;
  float* rowm = sm + 3 * GS;
  float* mDb  = sm + 4 * GS;
  float* hs   = xm;             // h_state aliases x_mlstm (dead after attn)

  const int M = BB * SSEQ;      // 4096

  // 1) up projection: x @ W_up -> [x_mlstm | z]
  gemm_bf16_kernel<<<dim3((2 * INNER) / 128, M / 128), 256, 0, stream>>>(
      x, W_up, xm, zb, M, EE, 2 * INNER, INNER);

  // 2) conv + swish + headwise q/k/v + gate preactivations
  pre_kernel<<<dim3(M), 256, 0, stream>>>(xm, conv_w, conv_b, Wq, Wk, Wv,
                                          W_ig, b_ig, W_fg, b_fg,
                                          xa, qb, kb, vt, igb, fgb);

  // 3) per-(b,nh) gate scan
  scan_kernel<<<dim3(BB * NHEAD), 1, 0, stream>>>(igb, fgb, gcol, rowm, mDb);

  // 4) streaming mLSTM attention (WMMA bf16)
  attn_kernel<<<dim3(SSEQ / 32, NHEAD, BB), 256, 0, stream>>>(
      qb, kb, vt, gcol, rowm, mDb, hb);

  // 5) multi-head LN + skip + swish(z) gate
  post_kernel<<<dim3(M), 256, 0, stream>>>(hb, xa, zb, norm_w, skip, hs);

  // 6) down projection -> output
  gemm_bf16_kernel<<<dim3(EE / 128, M / 128), 256, 0, stream>>>(
      hs, W_down, out, out, M, INNER, EE, EE);
}
